// GCNN_RNN_90744069030214
// MI455X (gfx1250) — matrix operand order repaired and checked by reading, not compiled, not run on hardware
//
#include <hip/hip_runtime.h>
#include <hip/hip_bf16.h>
#include <math.h>

typedef __bf16 bf16;
typedef __attribute__((ext_vector_type(16))) __bf16 v16bf;
typedef __attribute__((ext_vector_type(8)))  float  v8f;

#define NN    5000
#define EE    80000
#define BB    8
#define TIN   12
#define TPRED 6
#define FF    8
#define OUTC  16
#define HH    64
#define DD    24
#define BTT   (BB*TIN)      // 96
#define BNN   (BB*NN)       // 40000
#define TTOT  (TIN+TPRED)   // 18

// packed-weight tile indices (each tile = 512 bf16 = 32 lanes * 16 halves = 1KB)
#define T_WIH 0     // 4 tiles  : W_ih^T  (K=24 pad 32, N=64)          idx = nt
#define T_WHH 4     // 8 tiles  : W_hh^T  (K=64, N=64)                 idx = kt*4+nt
#define T_W1  12    // 32 tiles : W1      (K=64, N=256)                idx = kt*16+nt
#define T_W2  44    // 128 tiles: W2      (K=256,N=256)                idx = kt*16+nt
#define T_W3  172   // 16 tiles : W3      (K=256,N=24 pad 32)          idx = kt*2+nt
#define NTILES 188

// bias LDS layout (floats)
#define BO_RNN 0    // 64  : b_ih + b_hh
#define BO_B1  64   // 256
#define BO_B2  320  // 256
#define BO_B3  576  // 24
#define BSZ    608

union AFrag { v16bf v; float4 f4[2]; bf16 h[16]; };

__device__ __forceinline__ v8f wmma_bf16(v16bf a, v16bf b, v8f c) {
  return __builtin_amdgcn_wmma_f32_16x16x32_bf16(false, a, false, b, (short)0, c, false, false);
}

// branchless tanh: 1 - 2/(e^{2x}+1); correct limits at +/-inf, no divergence
__device__ __forceinline__ float fast_tanh(float x) {
  float e = __expf(2.0f * x);
  return 1.0f - 2.0f * __builtin_amdgcn_rcpf(e + 1.0f);
}

// A fragment (16x32 bf16) from a row-major bf16 LDS buffer.
__device__ __forceinline__ v16bf load_a_lds(const bf16* rowbase, int kbase, int group) {
  AFrag a;
  const bf16* p = rowbase + kbase + group * 8;
  a.f4[0] = *(const float4*)(p);
  a.f4[1] = *(const float4*)(p + 16);
  return a.v;
}

// B fragment (32x16 bf16) from LDS-staged packed weights: 32 contiguous bytes per lane.
__device__ __forceinline__ v16bf load_b_lds(const bf16* wlds, int tile, int lane) {
  AFrag b;
  const bf16* p = wlds + tile * 512 + lane * 16;
  b.f4[0] = *(const float4*)(p);
  b.f4[1] = *(const float4*)(p + 8);
  return b.v;
}

// ---------------- GCN pipeline ----------------

__global__ void k_zero(float* deg, int* cnt, int* cur) {
  int i = blockIdx.x * blockDim.x + threadIdx.x;
  if (i < NN) { deg[i] = 0.f; cnt[i] = 0; cur[i] = 0; }
}

__global__ void k_deg(const int* __restrict__ src, const int* __restrict__ dst,
                      const float* __restrict__ ew, float* deg, int* cnt) {
  int i = blockIdx.x * blockDim.x + threadIdx.x;
  if (i < EE) {
    int d = dst[i];
    atomicAdd(&deg[d], ew[i]);
    atomicAdd(&cnt[d], 1);
  } else if (i < EE + NN) {
    atomicAdd(&deg[i - EE], 1.0f);   // self-loop weight 1
  }
}

__global__ void k_dinv(const float* __restrict__ deg, float* __restrict__ dinv) {
  int i = blockIdx.x * blockDim.x + threadIdx.x;
  if (i < NN) dinv[i] = deg[i] > 0.f ? 1.0f / sqrtf(deg[i]) : 0.f;
}

// single-block exclusive scan of cnt[0..NN) -> rowstart[0..NN]
__global__ void k_scan(const int* __restrict__ cnt, int* __restrict__ rowstart) {
  __shared__ int part[1024];
  __shared__ int base[1024];
  int tid = threadIdx.x;
  int c[5]; int sum = 0;
#pragma unroll
  for (int j = 0; j < 5; ++j) {
    int idx = tid * 5 + j;
    c[j] = idx < NN ? cnt[idx] : 0;
    sum += c[j];
  }
  part[tid] = sum;
  __syncthreads();
  if (tid == 0) {
    int acc = 0;
    for (int i = 0; i < 1024; ++i) { base[i] = acc; acc += part[i]; }
    rowstart[NN] = acc;
  }
  __syncthreads();
  int acc = base[tid];
#pragma unroll
  for (int j = 0; j < 5; ++j) {
    int idx = tid * 5 + j;
    if (idx < NN) rowstart[idx] = acc;
    acc += c[j];
  }
}

__global__ void k_fill(const int* __restrict__ src, const int* __restrict__ dst,
                       const float* __restrict__ ew, const float* __restrict__ dinv,
                       const int* __restrict__ rowstart, int* cur,
                       int* __restrict__ csr_src, float* __restrict__ csr_w) {
  int e = blockIdx.x * blockDim.x + threadIdx.x;
  if (e >= EE) return;
  int d = dst[e];
  int pos = atomicAdd(&cur[d], 1);
  int j = rowstart[d] + pos;
  int s = src[e];
  csr_src[j] = s;
  csr_w[j] = ew[e] * dinv[s];      // dst factor applied in k_agg
}

__global__ void k_xl(const float* __restrict__ x, const float* __restrict__ gW,
                     float* __restrict__ xl) {
  int r = blockIdx.x * blockDim.x + threadIdx.x;
  if (r >= BTT * NN) return;
  float xi[8];
  const float4* p = (const float4*)(x + (size_t)r * 8);
  ((float4*)xi)[0] = p[0]; ((float4*)xi)[1] = p[1];
  float o[16];
#pragma unroll
  for (int c = 0; c < 16; ++c) {
    float s = 0.f;
#pragma unroll
    for (int f = 0; f < 8; ++f) s += xi[f] * gW[f * 16 + c];
    o[c] = s;
  }
  float4* q = (float4*)(xl + (size_t)r * 16);
  q[0] = ((float4*)o)[0]; q[1] = ((float4*)o)[1];
  q[2] = ((float4*)o)[2]; q[3] = ((float4*)o)[3];
}

__global__ void k_agg(const float* __restrict__ xl, const int* __restrict__ rowstart,
                      const int* __restrict__ csr_src, const float* __restrict__ csr_w,
                      const float* __restrict__ dinv, const float* __restrict__ gb,
                      float* __restrict__ agg) {
  int r = blockIdx.x * blockDim.x + threadIdx.x;
  if (r >= BTT * NN) return;
  int bt = r / NN, n = r - bt * NN;
  float di = dinv[n];
  float a[16];
#pragma unroll
  for (int c = 0; c < 16; ++c) a[c] = di * xl[(size_t)r * 16 + c];   // self-loop term
  const float* xb = xl + (size_t)bt * NN * 16;
  int s0 = rowstart[n], s1 = rowstart[n + 1];
  for (int j = s0; j < s1; ++j) {
    int s = csr_src[j];
    float wn = csr_w[j];
    const float* xs = xb + (size_t)s * 16;
#pragma unroll
    for (int c = 0; c < 16; ++c) a[c] += wn * xs[c];
  }
#pragma unroll
  for (int c = 0; c < 16; ++c) agg[(size_t)r * 16 + c] = gb[c] + di * a[c];
}

// ---------------- weight packing into WMMA B-fragment order ----------------
__global__ void k_packW(const float* __restrict__ Wih, const float* __restrict__ Whh,
                        const float* __restrict__ W1, const float* __restrict__ W2,
                        const float* __restrict__ W3, bf16* __restrict__ packW) {
  int g = blockIdx.x * blockDim.x + threadIdx.x;
  if (g >= NTILES * 512) return;
  int tile = g >> 9;
  int r = g & 511;
  int lane = r >> 4;
  int idx = r & 15;
  int nl = lane & 15, group = lane >> 4;
  int kt_k = (idx < 8) ? (group * 8 + idx) : (16 + group * 8 + (idx - 8));   // k within tile (0..31)
  float v = 0.f;
  if (tile < T_WHH) {                    // W_ih^T : B[k][n] = Wih[n][k], Wih is [64][24]
    int nt = tile - T_WIH;
    int k = kt_k, n = nt * 16 + nl;
    if (k < DD) v = Wih[n * DD + k];
  } else if (tile < T_W1) {              // W_hh^T : B[k][n] = Whh[n][k], Whh is [64][64]
    int ti = tile - T_WHH;
    int kt = ti >> 2, nt = ti & 3;
    int k = kt * 32 + kt_k, n = nt * 16 + nl;
    v = Whh[n * HH + k];
  } else if (tile < T_W2) {              // W1 [64][256]
    int ti = tile - T_W1;
    int kt = ti >> 4, nt = ti & 15;
    int k = kt * 32 + kt_k, n = nt * 16 + nl;
    v = W1[k * 256 + n];
  } else if (tile < T_W3) {              // W2 [256][256]
    int ti = tile - T_W2;
    int kt = ti >> 4, nt = ti & 15;
    int k = kt * 32 + kt_k, n = nt * 16 + nl;
    v = W2[k * 256 + n];
  } else {                               // W3 [256][24] pad N to 32
    int ti = tile - T_W3;
    int kt = ti >> 1, nt = ti & 1;
    int k = kt * 32 + kt_k, n = nt * 16 + nl;
    if (n < DD) v = W3[k * DD + n];
  }
  packW[g] = (bf16)v;
}

// ---------------- fused RNN + MLP (all WMMA, weights + biases LDS-resident) ----------------

__device__ __forceinline__ void mlp_out(
    const bf16* wlds, const float* blds, const bf16* hb, bf16* ab,
    float* __restrict__ out, int bn0, int t_out, int lane) {
  const int nl = lane & 15, group = lane >> 4;
  // layer 1: 64 -> 256, ReLU  (2 N-tiles / iter, independent accumulators)
  v16bf g0 = load_a_lds(hb + nl * 64, 0, group);
  v16bf g1 = load_a_lds(hb + nl * 64, 32, group);
#pragma unroll 1
  for (int nt2 = 0; nt2 < 8; ++nt2) {
    const int nt0 = nt2 * 2, nt1 = nt0 + 1;
    v8f acc0 = {}, acc1 = {};
    acc0 = wmma_bf16(g0, load_b_lds(wlds, T_W1 + nt0, lane), acc0);
    acc1 = wmma_bf16(g0, load_b_lds(wlds, T_W1 + nt1, lane), acc1);
    acc0 = wmma_bf16(g1, load_b_lds(wlds, T_W1 + 16 + nt0, lane), acc0);
    acc1 = wmma_bf16(g1, load_b_lds(wlds, T_W1 + 16 + nt1, lane), acc1);
    float bias0 = blds[BO_B1 + nt0 * 16 + nl];
    float bias1 = blds[BO_B1 + nt1 * 16 + nl];
#pragma unroll
    for (int v = 0; v < 8; ++v) {
      float x0 = acc0[v] + bias0; x0 = x0 > 0.f ? x0 : 0.f;
      float x1 = acc1[v] + bias1; x1 = x1 > 0.f ? x1 : 0.f;
      ab[(group * 8 + v) * 256 + nt0 * 16 + nl] = (bf16)x0;
      ab[(group * 8 + v) * 256 + nt1 * 16 + nl] = (bf16)x1;
    }
  }
  __syncthreads();
  // layer 2: 256 -> 256, ReLU (wave's whole 16x256 A-tile in af[] -> in-place overwrite OK)
  v16bf af[8];
#pragma unroll
  for (int kt = 0; kt < 8; ++kt) af[kt] = load_a_lds(ab + nl * 256, kt * 32, group);
#pragma unroll 1
  for (int nt2 = 0; nt2 < 8; ++nt2) {
    const int nt0 = nt2 * 2, nt1 = nt0 + 1;
    v8f acc0 = {}, acc1 = {};
#pragma unroll
    for (int kt = 0; kt < 8; ++kt) {
      acc0 = wmma_bf16(af[kt], load_b_lds(wlds, T_W2 + kt * 16 + nt0, lane), acc0);
      acc1 = wmma_bf16(af[kt], load_b_lds(wlds, T_W2 + kt * 16 + nt1, lane), acc1);
    }
    float bias0 = blds[BO_B2 + nt0 * 16 + nl];
    float bias1 = blds[BO_B2 + nt1 * 16 + nl];
#pragma unroll
    for (int v = 0; v < 8; ++v) {
      float x0 = acc0[v] + bias0; x0 = x0 > 0.f ? x0 : 0.f;
      float x1 = acc1[v] + bias1; x1 = x1 > 0.f ? x1 : 0.f;
      ab[(group * 8 + v) * 256 + nt0 * 16 + nl] = (bf16)x0;
      ab[(group * 8 + v) * 256 + nt1 * 16 + nl] = (bf16)x1;
    }
  }
  __syncthreads();
  // layer 3: 256 -> 24 (padded 32), both N-tiles interleaved, write to flat output view
#pragma unroll
  for (int kt = 0; kt < 8; ++kt) af[kt] = load_a_lds(ab + nl * 256, kt * 32, group);
  {
    v8f acc0 = {}, acc1 = {};
#pragma unroll
    for (int kt = 0; kt < 8; ++kt) {
      acc0 = wmma_bf16(af[kt], load_b_lds(wlds, T_W3 + kt * 2 + 0, lane), acc0);
      acc1 = wmma_bf16(af[kt], load_b_lds(wlds, T_W3 + kt * 2 + 1, lane), acc1);
    }
    // nt0: d = nl (always < 24)
    {
      float bias = blds[BO_B3 + nl];
#pragma unroll
      for (int v = 0; v < 8; ++v) {
        int rr = bn0 + group * 8 + v;
        if (rr < BNN)
          out[(size_t)rr * (TTOT * DD) + t_out * DD + nl] = acc0[v] + bias;
      }
    }
    // nt1: d = 16 + nl, valid only for nl < 8
    if (nl < 8) {
      int d = 16 + nl;
      float bias = blds[BO_B3 + d];
#pragma unroll
      for (int v = 0; v < 8; ++v) {
        int rr = bn0 + group * 8 + v;
        if (rr < BNN)
          out[(size_t)rr * (TTOT * DD) + t_out * DD + d] = acc1[v] + bias;
      }
    }
  }
  __syncthreads();
}

__global__ __launch_bounds__(256) void k_rnn_mlp(
    const float* __restrict__ x_in, const float* __restrict__ agg,
    const bf16* __restrict__ packW,
    const float* __restrict__ b_ih, const float* __restrict__ b_hh,
    const float* __restrict__ b1, const float* __restrict__ b2, const float* __restrict__ b3,
    float* __restrict__ out) {
  __shared__ __align__(16) bf16 wlds[NTILES * 512];   // 188KB packed weights
  __shared__ __align__(16) bf16 hbuf[128 * 64];       // 16KB hidden states (bf16)
  __shared__ __align__(16) bf16 abuf[128 * 256];      // 64KB activations (in-place L1/L2)
  __shared__ __align__(16) float blds[BSZ];           // 2.4KB biases

  const int tid = threadIdx.x;
  const int lane = tid & 31;
  const int wave = tid >> 5;
  const int nl = lane & 15;
  const int group = lane >> 4;
  const int row0 = wave * 16;                   // wave-private 16-row tile
  const int bn0 = blockIdx.x * 128 + row0;      // global sequence row of tile row 0
  int bnl = bn0 + nl;                           // row this lane gathers for A
  if (bnl >= BNN) bnl = BNN - 1;                // clamp tail (stores are guarded)

  // stage packed weights + biases into LDS (once per block), zero h
  {
    const uint4* s = (const uint4*)packW;
    uint4* d = (uint4*)wlds;
    for (int i = tid; i < NTILES * 512 / 8; i += 256) d[i] = s[i];
  }
  for (int i = tid; i < 128 * 64; i += 256) hbuf[i] = (bf16)0.f;
  if (tid < 64) blds[BO_RNN + tid] = b_ih[tid] + b_hh[tid];
  blds[BO_B1 + tid] = b1[tid];
  blds[BO_B2 + tid] = b2[tid];
  if (tid < 24) blds[BO_B3 + tid] = b3[tid];
  __syncthreads();

  bf16* hb = hbuf + row0 * 64;
  bf16* ab = abuf + row0 * 256;

#pragma unroll 1
  for (int t = 0; t < TIN; ++t) {
    // gather this row's step-t input per the flat .view semantics (contiguous chunks)
    float xr[24];
    {
      float4* xf = (float4*)xr;
      const float4* px = (const float4*)(x_in + (size_t)bnl * (TIN * FF) + t * FF);
      xf[0] = px[0]; xf[1] = px[1];
      const float4* pa = (const float4*)(agg + (size_t)bnl * (TIN * OUTC) + t * OUTC);
      xf[2] = pa[0]; xf[3] = pa[1]; xf[4] = pa[2]; xf[5] = pa[3];
    }
    AFrag xa;
#pragma unroll
    for (int j = 0; j < 8; ++j) {
      int kk = ((j < 4) ? (2 * j) : (16 + 2 * (j - 4))) + group * 8;
      xa.h[2 * j]     = (kk     < DD) ? (bf16)xr[kk]     : (bf16)0.f;
      xa.h[2 * j + 1] = (kk + 1 < DD) ? (bf16)xr[kk + 1] : (bf16)0.f;
    }
    v16bf h0 = load_a_lds(hb + nl * 64, 0, group);
    v16bf h1 = load_a_lds(hb + nl * 64, 32, group);
#pragma unroll 1
    for (int p = 0; p < 2; ++p) {
      const int nt0 = p * 2, nt1 = nt0 + 1;
      v8f acc0 = {}, acc1 = {};
      acc0 = wmma_bf16(xa.v, load_b_lds(wlds, T_WIH + nt0, lane), acc0);
      acc1 = wmma_bf16(xa.v, load_b_lds(wlds, T_WIH + nt1, lane), acc1);
      acc0 = wmma_bf16(h0, load_b_lds(wlds, T_WHH + nt0, lane), acc0);
      acc1 = wmma_bf16(h0, load_b_lds(wlds, T_WHH + nt1, lane), acc1);
      acc0 = wmma_bf16(h1, load_b_lds(wlds, T_WHH + 4 + nt0, lane), acc0);
      acc1 = wmma_bf16(h1, load_b_lds(wlds, T_WHH + 4 + nt1, lane), acc1);
      float bias0 = blds[BO_RNN + nt0 * 16 + nl];
      float bias1 = blds[BO_RNN + nt1 * 16 + nl];
#pragma unroll
      for (int v = 0; v < 8; ++v) {
        hb[(group * 8 + v) * 64 + nt0 * 16 + nl] = (bf16)fast_tanh(acc0[v] + bias0);
        hb[(group * 8 + v) * 64 + nt1 * 16 + nl] = (bf16)fast_tanh(acc1[v] + bias1);
      }
    }
    __syncthreads();
    mlp_out(wlds, blds, hb, ab, out, bn0, t, lane);
  }

#pragma unroll 1
  for (int s = 0; s < TPRED; ++s) {
    v16bf h0 = load_a_lds(hb + nl * 64, 0, group);
    v16bf h1 = load_a_lds(hb + nl * 64, 32, group);
#pragma unroll 1
    for (int p = 0; p < 2; ++p) {
      const int nt0 = p * 2, nt1 = nt0 + 1;
      v8f acc0 = {}, acc1 = {};
      acc0 = wmma_bf16(h0, load_b_lds(wlds, T_WHH + nt0, lane), acc0);
      acc1 = wmma_bf16(h0, load_b_lds(wlds, T_WHH + nt1, lane), acc1);
      acc0 = wmma_bf16(h1, load_b_lds(wlds, T_WHH + 4 + nt0, lane), acc0);
      acc1 = wmma_bf16(h1, load_b_lds(wlds, T_WHH + 4 + nt1, lane), acc1);
      float bias0 = blds[BO_RNN + nt0 * 16 + nl];
      float bias1 = blds[BO_RNN + nt1 * 16 + nl];
#pragma unroll
      for (int v = 0; v < 8; ++v) {
        hb[(group * 8 + v) * 64 + nt0 * 16 + nl] = (bf16)fast_tanh(acc0[v] + bias0);
        hb[(group * 8 + v) * 64 + nt1 * 16 + nl] = (bf16)fast_tanh(acc1[v] + bias1);
      }
    }
    __syncthreads();
    mlp_out(wlds, blds, hb, ab, out, bn0, TIN + s, lane);
  }
}

// ---------------- host launcher ----------------

extern "C" void kernel_launch(void* const* d_in, const int* in_sizes, int n_in,
                              void* d_out, int out_size, void* d_ws, size_t ws_size,
                              hipStream_t stream) {
  const float* x_in  = (const float*)d_in[0];
  const int*   eidx  = (const int*)d_in[1];
  const float* ew    = (const float*)d_in[2];
  const float* gcn_W = (const float*)d_in[3];
  const float* gcn_b = (const float*)d_in[4];
  const float* W_ih  = (const float*)d_in[5];
  const float* W_hh  = (const float*)d_in[6];
  const float* b_ih  = (const float*)d_in[7];
  const float* b_hh  = (const float*)d_in[8];
  const float* W1    = (const float*)d_in[9];
  const float* b1    = (const float*)d_in[10];
  const float* W2    = (const float*)d_in[11];
  const float* b2    = (const float*)d_in[12];
  const float* W3    = (const float*)d_in[13];
  const float* b3    = (const float*)d_in[14];
  // d_in[15] = pred_hor == 6 (compile-time constant here)
  const int* src = eidx;
  const int* dst = eidx + EE;

  char* w = (char*)d_ws;
  float* deg      = (float*)(w + 0);         // 20000 B
  float* dinv     = (float*)(w + 20480);     // 20000 B
  int*   cnt      = (int*)  (w + 40960);     // 20000 B
  int*   cur      = (int*)  (w + 61440);     // 20000 B
  int*   rowstart = (int*)  (w + 81920);     // 20004 B
  int*   csr_src  = (int*)  (w + 102400);    // 320000 B
  float* csr_w    = (float*)(w + 422912);    // 320000 B
  float* xl       = (float*)(w + 743424);    // 30,720,000 B
  float* agg      = (float*)(w + 31463424);  // 30,720,000 B
  bf16*  packW    = (bf16*) (w + 62183424);  // 192,512 B

  k_zero<<<(NN + 255) / 256, 256, 0, stream>>>(deg, cnt, cur);
  k_deg<<<(EE + NN + 255) / 256, 256, 0, stream>>>(src, dst, ew, deg, cnt);
  k_dinv<<<(NN + 255) / 256, 256, 0, stream>>>(deg, dinv);
  k_scan<<<1, 1024, 0, stream>>>(cnt, rowstart);
  k_fill<<<(EE + 255) / 256, 256, 0, stream>>>(src, dst, ew, dinv, rowstart, cur, csr_src, csr_w);
  k_xl<<<(BTT * NN + 255) / 256, 256, 0, stream>>>(x_in, gcn_W, xl);
  k_agg<<<(BTT * NN + 255) / 256, 256, 0, stream>>>(xl, rowstart, csr_src, csr_w, dinv, gcn_b, agg);
  k_packW<<<(NTILES * 512 + 255) / 256, 256, 0, stream>>>(W_ih, W_hh, W1, W2, W3, packW);
  k_rnn_mlp<<<(BNN + 127) / 128, 256, 0, stream>>>(x_in, agg, packW, b_ih, b_hh,
                                                   b1, b2, b3, (float*)d_out);
}
